// SpikeRNN_90795608637854
// MI455X (gfx1250) — compile-verified, matrix operand order checked by reading, not run on hardware
//
#include <hip/hip_runtime.h>

// ---------------------------------------------------------------------------
// SpikeRNN forward, fused WMMA-GEMM + LIF for gfx1250 (MI455X, wave32, WMMA)
//   T=4, TAU=2 (charge: v=(v+x)/2), V_TH=1, hard reset to 0
//   B=16, L=512, C=128, D=1024, M=B*L=8192
// ---------------------------------------------------------------------------

typedef __attribute__((ext_vector_type(16))) _Float16 v16h;
typedef __attribute__((ext_vector_type(8)))  _Float16 v8h;
typedef __attribute__((ext_vector_type(8)))  float    v8f;

#define T_STEPS 4
#define B_DIM   16
#define L_DIM   512
#define C_DIM   128
#define D_DIM   1024
#define M_ROWS  (B_DIM * L_DIM)   // 8192

__device__ __forceinline__ v16h join16(v8h lo, v8h hi) {
  return __builtin_shufflevector(lo, hi, 0, 1, 2, 3, 4, 5, 6, 7, 8, 9, 10, 11,
                                 12, 13, 14, 15);
}

// ---------------------------------------------------------------------------
// Repack row-major f32 weights [K][N] into f16 WMMA B-fragment order.
// Fragment layout (per 32K x 16N tile): lane = hi*16 + n_local,
//   half j in 0..7  -> K = kb*32 + hi*8 + j
//   half j in 8..15 -> K = kb*32 + 16 + hi*8 + (j-8)
// Each lane's 16 halves contiguous at ((kb*(N/16) + nb)*32 + lane)*16.
// ---------------------------------------------------------------------------
__global__ void pack_w(const float* __restrict__ W, _Float16* __restrict__ out,
                       int K, int N) {
  int tid = blockIdx.x * blockDim.x + threadIdx.x;
  if (tid >= K * N) return;
  int k = tid / N, n = tid - k * N;
  int kb = k >> 5, dk = k & 31;
  int nb = n >> 4, nl = n & 15;
  int hi = (dk >> 3) & 1;
  int j  = (dk & 7) + ((dk >= 16) ? 8 : 0);
  int lane = hi * 16 + nl;
  size_t idx = ((((size_t)kb * (N >> 4)) + nb) * 32 + lane) * 16 + j;
  out[idx] = (_Float16)W[(size_t)k * N + n];
}

// ---------------------------------------------------------------------------
// Stage 0: elementwise LIF over T on the raw input (input constant across T).
// Materializes s0 [T, M, C] f16 (8 MB, L2-resident) exactly once.
// ---------------------------------------------------------------------------
__global__ __launch_bounds__(256)
void spike_encode(const float* __restrict__ X, _Float16* __restrict__ S0) {
  const int i = blockIdx.x * 256 + threadIdx.x;  // over M*C
  const float x = X[i];
  float v = 0.f;
#pragma unroll
  for (int t = 0; t < T_STEPS; ++t) {
    v = 0.5f * (v + x);
    const bool s = (v >= 1.0f);
    S0[(size_t)t * (M_ROWS * C_DIM) + i] = s ? (_Float16)1.0f : (_Float16)0.0f;
    v = s ? 0.0f : v;
  }
}

// ---------------------------------------------------------------------------
// Unified spike GEMM + fused LIF-over-T epilogue.
//   A: [T, M, K_DIM] f16 spikes (exact {0,1}), Bp: packed f16 [K_DIM x D]
//   8 waves/block; wave tile 32 rows x 32 cols (2 M-frags x 2 N-frags),
//   block tile 64 x 128; 16 accumulator chains (4 T x 2 x 2) per wave.
//   Per K-chunk: all 20 b128 loads issued up front, then 16 WMMAs.
// FINAL=false: write next spike buffer. FINAL=true: write mean-over-T (f32).
// ---------------------------------------------------------------------------
template <int K_DIM, bool FINAL>
__global__ __launch_bounds__(256)
void gemm_lif(const _Float16* __restrict__ A,    // [T, M, K_DIM]
              const _Float16* __restrict__ Bp,   // packed [K_DIM][D]
              const float* __restrict__ bias,    // [D]
              _Float16* __restrict__ Sout,       // [T, M, D] (if !FINAL)
              float* __restrict__ Out) {         // [M, D]    (if FINAL)
  const int lane = threadIdx.x & 31;
  const int wave = threadIdx.x >> 5;
  const int wm = wave >> 2;                 // 0..1
  const int wn = wave & 3;                  // 0..3
  const int m0 = blockIdx.y * 64 + wm * 32;
  const int nt0 = blockIdx.x * 8 + wn * 2;  // first of two 16-wide N tiles
  const int hi = lane >> 4;
  const int ml = lane & 15;

  v8f acc[T_STEPS][2][2] = {};  // [t][m-frag][n-frag]

  const _Float16* arow0 = A + (size_t)(m0 + ml) * K_DIM + hi * 8;
  const size_t tstride = (size_t)M_ROWS * K_DIM;

#pragma unroll 2
  for (int kb = 0; kb < K_DIM / 32; ++kb) {
    const int kk = kb * 32;
    // prefetch next K-chunk's B fragments (global_prefetch path)
    if (kb + 1 < K_DIM / 32) {
      __builtin_prefetch(
          Bp + ((((size_t)(kb + 1) * (D_DIM >> 4)) + nt0) * 32 + lane) * 16, 0,
          1);
    }
    // ---- issue ALL fragment loads for this K-chunk first ----
    v16h bfr[2];
#pragma unroll
    for (int un = 0; un < 2; ++un) {
      const _Float16* bptr =
          Bp + ((((size_t)kb * (D_DIM >> 4)) + nt0 + un) * 32 + lane) * 16;
      bfr[un] = join16(*(const v8h*)(bptr), *(const v8h*)(bptr + 8));
    }
    v16h afr[T_STEPS][2];
#pragma unroll
    for (int t = 0; t < T_STEPS; ++t) {
#pragma unroll
      for (int um = 0; um < 2; ++um) {
        // A layout is K-contiguous per lane: halves 0..7 at K=kk+hi*8,
        // halves 8..15 at K=kk+16+hi*8
        const _Float16* ar =
            arow0 + (size_t)t * tstride + (size_t)um * 16 * K_DIM + kk;
        afr[t][um] = join16(*(const v8h*)(ar), *(const v8h*)(ar + 16));
      }
    }
    // ---- then a dense run of 16 WMMAs ----
#pragma unroll
    for (int t = 0; t < T_STEPS; ++t) {
#pragma unroll
      for (int um = 0; um < 2; ++um) {
#pragma unroll
        for (int un = 0; un < 2; ++un) {
          acc[t][um][un] = __builtin_amdgcn_wmma_f32_16x16x32_f16(
              false, afr[t][um], false, bfr[un], (short)0, acc[t][um][un],
              false, false);
        }
      }
    }
  }

  // LIF-over-T epilogue; C/D layout: VGPR r, lane -> (m = r + 8*hi, n = ml)
#pragma unroll
  for (int um = 0; um < 2; ++um) {
#pragma unroll
    for (int un = 0; un < 2; ++un) {
      const int n = (nt0 + un) * 16 + ml;
      const float bv = bias[n];
#pragma unroll
      for (int r = 0; r < 8; ++r) {
        const int m = m0 + um * 16 + r + 8 * hi;
        float v = 0.f;
        float smean = 0.f;
#pragma unroll
        for (int t = 0; t < T_STEPS; ++t) {
          const float y = acc[t][um][un][r] + bv;
          v = 0.5f * (v + y);
          const bool s = (v >= 1.0f);
          v = s ? 0.0f : v;
          if constexpr (FINAL) {
            smean += s ? 1.0f : 0.0f;
          } else {
            Sout[((size_t)t * M_ROWS + m) * D_DIM + n] =
                s ? (_Float16)1.0f : (_Float16)0.0f;
          }
        }
        if constexpr (FINAL) {
          Out[(size_t)m * D_DIM + n] = smean * 0.25f;
        }
      }
    }
  }
}

// ---------------------------------------------------------------------------
// out2[b, d] = mean over l of out[b, l, d] (coalesced over d)
// ---------------------------------------------------------------------------
__global__ __launch_bounds__(256)
void row_mean(const float* __restrict__ out, float* __restrict__ out2) {
  const int d = blockIdx.x * blockDim.x + threadIdx.x;
  const int b = blockIdx.y;
  float s = 0.f;
  for (int l = 0; l < L_DIM; ++l)
    s += out[((size_t)b * L_DIM + l) * D_DIM + d];
  out2[(size_t)b * D_DIM + d] = s * (1.0f / (float)L_DIM);
}

// ---------------------------------------------------------------------------
extern "C" void kernel_launch(void* const* d_in, const int* in_sizes, int n_in,
                              void* d_out, int out_size, void* d_ws,
                              size_t ws_size, hipStream_t stream) {
  const float* inputs  = (const float*)d_in[0];  // [B, L, C]
  const float* enc_W   = (const float*)d_in[1];  // [C, D]
  const float* enc_b   = (const float*)d_in[2];  // [D]
  const float* W_cells = (const float*)d_in[3];  // [2, D, D]
  const float* b_cells = (const float*)d_in[4];  // [2, D]

  char* ws = (char*)d_ws;
  size_t off = 0;
  _Float16* encWp = (_Float16*)(ws + off); off += (size_t)C_DIM * D_DIM * 2;
  _Float16* W0p   = (_Float16*)(ws + off); off += (size_t)D_DIM * D_DIM * 2;
  _Float16* W1p   = (_Float16*)(ws + off); off += (size_t)D_DIM * D_DIM * 2;
  off = (off + 255) & ~(size_t)255;
  _Float16* s0 = (_Float16*)(ws + off);
  off += (size_t)T_STEPS * M_ROWS * C_DIM * 2;
  _Float16* s1 = (_Float16*)(ws + off);
  off += (size_t)T_STEPS * M_ROWS * D_DIM * 2;
  _Float16* s2 = (_Float16*)(ws + off);

  // Weight repack (deterministic, every call)
  pack_w<<<(C_DIM * D_DIM + 255) / 256, 256, 0, stream>>>(enc_W, encWp, C_DIM,
                                                          D_DIM);
  pack_w<<<(D_DIM * D_DIM + 255) / 256, 256, 0, stream>>>(W_cells, W0p, D_DIM,
                                                          D_DIM);
  pack_w<<<(D_DIM * D_DIM + 255) / 256, 256, 0, stream>>>(
      W_cells + (size_t)D_DIM * D_DIM, W1p, D_DIM, D_DIM);

  // Stage-0 spikes once (elementwise LIF)
  spike_encode<<<(M_ROWS * C_DIM) / 256, 256, 0, stream>>>(inputs, s0);

  dim3 grid(D_DIM / 128, M_ROWS / 64);  // (8, 128)
  gemm_lif<C_DIM, false><<<grid, 256, 0, stream>>>(s0, encWp, enc_b, s1,
                                                   nullptr);
  gemm_lif<D_DIM, false><<<grid, 256, 0, stream>>>(s1, W0p, b_cells, s2,
                                                   nullptr);
  gemm_lif<D_DIM, true><<<grid, 256, 0, stream>>>(s2, W1p, b_cells + D_DIM,
                                                  nullptr, (float*)d_out);

  float* out2 = (float*)d_out + (size_t)M_ROWS * D_DIM;
  row_mean<<<dim3(D_DIM / 256, B_DIM), 256, 0, stream>>>((const float*)d_out,
                                                         out2);
}
